// GlobalAttention_72335839199532
// MI455X (gfx1250) — matrix operand order repaired and checked by reading, not compiled, not run on hardware
//
#include <hip/hip_runtime.h>

// GlobalAttention (Luong general) for MI455X / gfx1250.
// wave32, bf16 WMMA (v_wmma_f32_16x16x32_bf16), async global->LDS staging
// (global_load_async_to_lds_b128 + s_wait_asynccnt), double-buffered tiles.
// B=32, T=512, S=512, D=1024.

constexpr int NB = 32;
constexpr int NT = 512;
constexpr int NS = 512;
constexpr int ND = 1024;

typedef __attribute__((ext_vector_type(16))) __bf16 v16bf;
typedef __attribute__((ext_vector_type(8)))  float  v8f;
typedef unsigned short u16;
typedef unsigned int   u32;

struct alignas(16) U4 { u32 x, y, z, w; };

constexpr int LDP = 40; // padded LDS row stride (elements) to spread banks

__device__ __forceinline__ u16 f2bf(float f) {
    union { float f; u32 u; } v; v.f = f;
    u32 r = v.u + 0x7FFFu + ((v.u >> 16) & 1u);   // round-to-nearest-even
    return (u16)(r >> 16);
}

union Frag { U4 u[2]; v16bf v; };

// A fragment, 16x32 bf16 tile, row-major in LDS (stride LDP elements).
// lane m (0-15), half h=lane>>4: dwords0-3 = K[h*8..+7], dwords4-7 = K[16+h*8..+7].
__device__ __forceinline__ v16bf ldsA(const u16* base, int lane) {
    int m = lane & 15, h = lane >> 4;
    const u16* p = base + m * LDP + h * 8;
    Frag r;
    r.u[0] = *(const U4*)(p);
    r.u[1] = *(const U4*)(p + 16);
    return r.v;
}

// B fragment, 32(K)x16(N) tile stored column-major in LDS as Bt[n][k].
// lane n (0-15), half h: dwords0-3 = K[h*8..+7], dwords4-7 = K[16+h*8..+7].
__device__ __forceinline__ v16bf ldsB(const u16* base, int lane) {
    int n = lane & 15, h = lane >> 4;
    const u16* p = base + n * LDP + h * 8;
    Frag r;
    r.u[0] = *(const U4*)(p);
    r.u[1] = *(const U4*)(p + 16);
    return r.v;
}

#define WMMA_BF16(a, b, c) \
    __builtin_amdgcn_wmma_f32_16x16x32_bf16(false, (a), false, (b), (short)0, (c), false, false)

// Async global->LDS 16B copy (per lane). GV mode: 64-bit global addr in VGPR
// pair, LDS byte address (workgroup-relative, = low 32 bits of flat shared
// address) in a VGPR. Tracked by ASYNCcnt.
__device__ __forceinline__ void async_copy16(u32 lds, const void* g) {
    asm volatile("global_load_async_to_lds_b128 %0, %1, off"
                 :: "v"(lds), "v"(g) : "memory");
}
#define WAIT_ASYNC(n) asm volatile("s_wait_asynccnt " #n ::: "memory")

__device__ __forceinline__ u32 lds_off(const void* p) {
    return (u32)(size_t)p;
}

// ---------------------------------------------------------------- convert
__global__ __launch_bounds__(256) void cvt_f32_bf16(const float* __restrict__ src,
                                                    u16* __restrict__ dst, long long n) {
    long long i = (long long)blockIdx.x * blockDim.x + threadIdx.x;
    long long stride = (long long)gridDim.x * blockDim.x;
    for (; i < n; i += stride) dst[i] = f2bf(src[i]);
}

// ------------------------------------------------- ht = q @ W_in^T  (bf16)
// grid (BT/128, D/128). Double-buffered async staging, 4 async ops/thread/iter.
__global__ __launch_bounds__(256) void ht_kernel(const u16* __restrict__ qB,
                                                 const u16* __restrict__ wInB,
                                                 u16* __restrict__ htB) {
    __shared__ __align__(16) u16 At[2][128 * LDP];
    __shared__ __align__(16) u16 Bt[2][128 * LDP];
    const int tid = threadIdx.x, lane = tid & 31, wid = tid >> 5;
    const int wm = wid & 3, wn = wid >> 2;          // 4x2 wave grid, wave tile 32x64
    const int row0 = blockIdx.x * 128, n0 = blockIdx.y * 128;
    const int r = tid >> 1, cc = (tid & 1) * 16;
    const u16* gA = qB + (size_t)(row0 + r) * ND + cc;
    const u16* gB = wInB + (size_t)(n0 + r) * ND + cc;
    u32 lA[2] = { lds_off(&At[0][r * LDP + cc]), lds_off(&At[1][r * LDP + cc]) };
    u32 lB[2] = { lds_off(&Bt[0][r * LDP + cc]), lds_off(&Bt[1][r * LDP + cc]) };
    v8f acc[2][4] = {};
    async_copy16(lA[0], gA);      async_copy16(lA[0] + 16, gA + 8);
    async_copy16(lB[0], gB);      async_copy16(lB[0] + 16, gB + 8);
    int kb = 0;
    for (int k0 = 0; k0 < ND; k0 += 32, kb ^= 1) {
        if (k0 + 32 < ND) {
            const u16* a = gA + k0 + 32;
            const u16* b = gB + k0 + 32;
            async_copy16(lA[kb ^ 1], a);      async_copy16(lA[kb ^ 1] + 16, a + 8);
            async_copy16(lB[kb ^ 1], b);      async_copy16(lB[kb ^ 1] + 16, b + 8);
            WAIT_ASYNC(4);
        } else {
            WAIT_ASYNC(0);
        }
        __syncthreads();
        v16bf a0 = ldsA(&At[kb][(wm * 32) * LDP], lane);
        v16bf a1 = ldsA(&At[kb][(wm * 32 + 16) * LDP], lane);
#pragma unroll
        for (int jn = 0; jn < 4; ++jn) {
            v16bf bf = ldsB(&Bt[kb][(wn * 64 + jn * 16) * LDP], lane);
            acc[0][jn] = WMMA_BF16(a0, bf, acc[0][jn]);
            acc[1][jn] = WMMA_BF16(a1, bf, acc[1][jn]);
        }
        __syncthreads();   // protect buffer reused by next iteration's copies
    }
    const int half = lane >> 4, nn = lane & 15;
#pragma unroll
    for (int im = 0; im < 2; ++im)
#pragma unroll
        for (int jn = 0; jn < 4; ++jn)
#pragma unroll
            for (int i = 0; i < 8; ++i) {
                int m = row0 + wm * 32 + im * 16 + half * 8 + i;
                int n = n0 + wn * 64 + jn * 16 + nn;
                htB[(size_t)m * ND + n] = f2bf(acc[im][jn][i]);
            }
}

// ----------------- align = ht @ mem^T, masked softmax over S (fused epilogue)
// grid (T/32, B). Block owns 32 full rows of S=512. 2x4 wave grid: wave 16x128.
__global__ __launch_bounds__(256) void align_kernel(const u16* __restrict__ htB,
                                                    const u16* __restrict__ mB,
                                                    const int* __restrict__ lens,
                                                    float* __restrict__ out_align,
                                                    u16* __restrict__ Pb) {
    __shared__ __align__(16) u16 At[32 * LDP];
    __shared__ __align__(16) u16 Bt[512 * LDP];
    __shared__ float red[2][16][4];
    const int tid = threadIdx.x, lane = tid & 31, wid = tid >> 5;
    const int wm = wid & 1, wn = wid >> 1;
    const int b = blockIdx.y, t0 = blockIdx.x * 32;
    const int len = lens[b];
    const u16* Ag = htB + ((size_t)b * NT + t0) * ND;
    const u16* Bg = mB + (size_t)b * NS * ND;
    v8f acc[8] = {};
    for (int k0 = 0; k0 < ND; k0 += 32) {
        if (tid < 64) {
            int r = tid >> 1, c = (tid & 1) * 16;
            const U4* g = (const U4*)(Ag + (size_t)r * ND + k0 + c);
            *(U4*)&At[r * LDP + c]     = g[0];
            *(U4*)&At[r * LDP + c + 8] = g[1];
        }
#pragma unroll
        for (int j = 0; j < 4; ++j) {
            int slot = tid + 256 * j;
            int r = slot >> 1, c = (slot & 1) * 16;
            const U4* g = (const U4*)(Bg + (size_t)r * ND + k0 + c);
            *(U4*)&Bt[r * LDP + c]     = g[0];
            *(U4*)&Bt[r * LDP + c + 8] = g[1];
        }
        __syncthreads();
        v16bf a = ldsA(At + (wm * 16) * LDP, lane);
#pragma unroll
        for (int jn = 0; jn < 8; ++jn) {
            v16bf bf = ldsB(Bt + (wn * 128 + jn * 16) * LDP, lane);
            acc[jn] = WMMA_BF16(a, bf, acc[jn]);
        }
        __syncthreads();
    }
    const int half = lane >> 4, nn = lane & 15;
#pragma unroll
    for (int jn = 0; jn < 8; ++jn) {
        int s = wn * 128 + jn * 16 + nn;
        if (s >= len)
#pragma unroll
            for (int i = 0; i < 8; ++i) acc[jn][i] = -__builtin_inff();
    }
    float mx[8], sm[8];
#pragma unroll
    for (int i = 0; i < 8; ++i) {
        float m = acc[0][i];
#pragma unroll
        for (int jn = 1; jn < 8; ++jn) m = fmaxf(m, acc[jn][i]);
        for (int off = 1; off < 16; off <<= 1) m = fmaxf(m, __shfl_xor(m, off, 32));
        mx[i] = m;
    }
    if (nn == 0)
#pragma unroll
        for (int i = 0; i < 8; ++i) red[wm][half * 8 + i][wn] = mx[i];
    __syncthreads();
#pragma unroll
    for (int i = 0; i < 8; ++i) {
        float m = red[wm][half * 8 + i][0];
        for (int w = 1; w < 4; ++w) m = fmaxf(m, red[wm][half * 8 + i][w]);
        mx[i] = m;
    }
    __syncthreads();
#pragma unroll
    for (int i = 0; i < 8; ++i) {
        float s = 0.f;
#pragma unroll
        for (int jn = 0; jn < 8; ++jn) {
            float p = __expf(acc[jn][i] - mx[i]);
            acc[jn][i] = p;
            s += p;
        }
        for (int off = 1; off < 16; off <<= 1) s += __shfl_xor(s, off, 32);
        sm[i] = s;
    }
    if (nn == 0)
#pragma unroll
        for (int i = 0; i < 8; ++i) red[wm][half * 8 + i][wn] = sm[i];
    __syncthreads();
#pragma unroll
    for (int i = 0; i < 8; ++i) {
        float s = red[wm][half * 8 + i][0] + red[wm][half * 8 + i][1] +
                  red[wm][half * 8 + i][2] + red[wm][half * 8 + i][3];
        sm[i] = 1.f / s;
    }
#pragma unroll
    for (int i = 0; i < 8; ++i) {
        int t = t0 + wm * 16 + half * 8 + i;
        float inv = sm[i];
#pragma unroll
        for (int jn = 0; jn < 8; ++jn) {
            int s = wn * 128 + jn * 16 + nn;
            float p = acc[jn][i] * inv;
            out_align[(size_t)t * (NB * NS) + (size_t)b * NS + s] = p; // [T,B,S]
            Pb[((size_t)b * NT + t) * NS + s] = f2bf(p);
        }
    }
}

// ----------------------------------------------- c = P @ mem  (bf16 out)
// grid (T/128, D/128, B). A async double-buffered; mem tile transposed while
// register-staging into LDS (K = s, so native layout is wrong-major).
__global__ __launch_bounds__(256) void ctx_kernel(const u16* __restrict__ Pb,
                                                  const u16* __restrict__ mB,
                                                  u16* __restrict__ cB) {
    __shared__ __align__(16) u16 At[2][128 * LDP];
    __shared__ __align__(16) u16 Bt[2][128 * LDP];
    const int tid = threadIdx.x, lane = tid & 31, wid = tid >> 5;
    const int wm = wid & 3, wn = wid >> 2;
    const int b = blockIdx.z, row0 = blockIdx.x * 128, n0 = blockIdx.y * 128;
    const int r = tid >> 1, cc = (tid & 1) * 16;
    const u16* gA = Pb + ((size_t)b * NT + row0 + r) * NS + cc;
    const int dg = tid & 15, sp = tid >> 4;
    const u16* gM = mB + (size_t)b * NS * ND + n0 + dg * 8;
    u32 lA[2] = { lds_off(&At[0][r * LDP + cc]), lds_off(&At[1][r * LDP + cc]) };
    v8f acc[2][4] = {};

    // transpose-stage a 32(K=s) x 128(N=d) mem tile into Bt[buf][d][s]
    auto stageB = [&](int buf, int k) {
        union { U4 u; u16 s[8]; } ra, rb;
        ra.u = *(const U4*)(gM + (size_t)(k + 2 * sp) * ND);
        rb.u = *(const U4*)(gM + (size_t)(k + 2 * sp + 1) * ND);
#pragma unroll
        for (int j = 0; j < 8; ++j) {
            u32 packed = (u32)ra.s[j] | ((u32)rb.s[j] << 16);
            *(u32*)&Bt[buf][(dg * 8 + j) * LDP + 2 * sp] = packed;
        }
    };

    async_copy16(lA[0], gA);      async_copy16(lA[0] + 16, gA + 8);
    stageB(0, 0);
    int kb = 0;
    for (int k0 = 0; k0 < NS; k0 += 32, kb ^= 1) {
        if (k0 + 32 < NS) {
            const u16* a = gA + k0 + 32;
            async_copy16(lA[kb ^ 1], a);      async_copy16(lA[kb ^ 1] + 16, a + 8);
            stageB(kb ^ 1, k0 + 32);
            WAIT_ASYNC(2);
        } else {
            WAIT_ASYNC(0);
        }
        __syncthreads();
        v16bf a0 = ldsA(&At[kb][(wm * 32) * LDP], lane);
        v16bf a1 = ldsA(&At[kb][(wm * 32 + 16) * LDP], lane);
#pragma unroll
        for (int jn = 0; jn < 4; ++jn) {
            v16bf bf = ldsB(&Bt[kb][(wn * 64 + jn * 16) * LDP], lane);
            acc[0][jn] = WMMA_BF16(a0, bf, acc[0][jn]);
            acc[1][jn] = WMMA_BF16(a1, bf, acc[1][jn]);
        }
        __syncthreads();
    }
    const int half = lane >> 4, nn = lane & 15;
#pragma unroll
    for (int im = 0; im < 2; ++im)
#pragma unroll
        for (int jn = 0; jn < 4; ++jn)
#pragma unroll
            for (int i = 0; i < 8; ++i) {
                int m = row0 + wm * 32 + im * 16 + half * 8 + i;
                int n = n0 + wn * 64 + jn * 16 + nn;
                cB[((size_t)b * NT + m) * ND + n] = f2bf(acc[im][jn][i]);
            }
}

// ---------------- attn_h = tanh([c,q] @ W_out^T), scatter to [T,B,D]
// grid (BT/128, D/128). K loops over 2D; A-source switches c -> q at k=D.
__global__ __launch_bounds__(256) void outproj_kernel(const u16* __restrict__ cB,
                                                      const u16* __restrict__ qB,
                                                      const u16* __restrict__ wOutB,
                                                      float* __restrict__ out_attn) {
    __shared__ __align__(16) u16 At[2][128 * LDP];
    __shared__ __align__(16) u16 Bt[2][128 * LDP];
    const int tid = threadIdx.x, lane = tid & 31, wid = tid >> 5;
    const int wm = wid & 3, wn = wid >> 2;
    const int row0 = blockIdx.x * 128, n0 = blockIdx.y * 128;
    const int r = tid >> 1, cc = (tid & 1) * 16;
    const u16* gC = cB + (size_t)(row0 + r) * ND + cc;
    const u16* gQ = qB + (size_t)(row0 + r) * ND + cc;
    const u16* gB = wOutB + (size_t)(n0 + r) * (2 * ND) + cc;
    u32 lA[2] = { lds_off(&At[0][r * LDP + cc]), lds_off(&At[1][r * LDP + cc]) };
    u32 lB[2] = { lds_off(&Bt[0][r * LDP + cc]), lds_off(&Bt[1][r * LDP + cc]) };
    v8f acc[2][4] = {};
    async_copy16(lA[0], gC);      async_copy16(lA[0] + 16, gC + 8);
    async_copy16(lB[0], gB);      async_copy16(lB[0] + 16, gB + 8);
    int kb = 0;
    for (int k0 = 0; k0 < 2 * ND; k0 += 32, kb ^= 1) {
        const int kn = k0 + 32;
        if (kn < 2 * ND) {
            const u16* a = (kn < ND) ? (gC + kn) : (gQ + (kn - ND));
            const u16* bsrc = gB + kn;
            async_copy16(lA[kb ^ 1], a);      async_copy16(lA[kb ^ 1] + 16, a + 8);
            async_copy16(lB[kb ^ 1], bsrc);   async_copy16(lB[kb ^ 1] + 16, bsrc + 8);
            WAIT_ASYNC(4);
        } else {
            WAIT_ASYNC(0);
        }
        __syncthreads();
        v16bf a0 = ldsA(&At[kb][(wm * 32) * LDP], lane);
        v16bf a1 = ldsA(&At[kb][(wm * 32 + 16) * LDP], lane);
#pragma unroll
        for (int jn = 0; jn < 4; ++jn) {
            v16bf bf = ldsB(&Bt[kb][(wn * 64 + jn * 16) * LDP], lane);
            acc[0][jn] = WMMA_BF16(a0, bf, acc[0][jn]);
            acc[1][jn] = WMMA_BF16(a1, bf, acc[1][jn]);
        }
        __syncthreads();
    }
    const int half = lane >> 4, nn = lane & 15;
#pragma unroll
    for (int im = 0; im < 2; ++im)
#pragma unroll
        for (int jn = 0; jn < 4; ++jn)
#pragma unroll
            for (int i = 0; i < 8; ++i) {
                int rg = row0 + wm * 32 + im * 16 + half * 8 + i; // = b*T + t
                int bb = rg >> 9, tt = rg & 511;
                int n = n0 + wn * 64 + jn * 16 + nn;
                out_attn[((size_t)tt * NB + bb) * ND + n] = tanhf(acc[im][jn][i]);
            }
}

extern "C" void kernel_launch(void* const* d_in, const int* in_sizes, int n_in,
                              void* d_out, int out_size, void* d_ws, size_t ws_size,
                              hipStream_t stream) {
    const float* q    = (const float*)d_in[0];   // [B,T,D]
    const float* mem  = (const float*)d_in[1];   // [B,S,D]
    const int*   lens = (const int*)d_in[2];     // [B]
    const float* Wi   = (const float*)d_in[3];   // [D,D]
    const float* Wo   = (const float*)d_in[4];   // [D,2D]

    float* out_attn  = (float*)d_out;                          // [T,B,D]
    float* out_align = out_attn + (size_t)NT * NB * ND;        // [T,B,S]

    char* w = (char*)d_ws;
    u16* qB    = (u16*)w; w += (size_t)NB * NT * ND * 2;
    u16* mB    = (u16*)w; w += (size_t)NB * NS * ND * 2;
    u16* wInB  = (u16*)w; w += (size_t)ND * ND * 2;
    u16* wOutB = (u16*)w; w += (size_t)ND * 2 * ND * 2;
    u16* htB   = (u16*)w; w += (size_t)NB * NT * ND * 2;
    u16* cB    = (u16*)w; w += (size_t)NB * NT * ND * 2;
    u16* Pb    = (u16*)w; w += (size_t)NB * NT * NS * 2;

    cvt_f32_bf16<<<2048, 256, 0, stream>>>(q,   qB,    (long long)NB * NT * ND);
    cvt_f32_bf16<<<2048, 256, 0, stream>>>(mem, mB,    (long long)NB * NS * ND);
    cvt_f32_bf16<<<512,  256, 0, stream>>>(Wi,  wInB,  (long long)ND * ND);
    cvt_f32_bf16<<<512,  256, 0, stream>>>(Wo,  wOutB, (long long)ND * 2 * ND);

    ht_kernel     <<<dim3((NB * NT) / 128, ND / 128), 256, 0, stream>>>(qB, wInB, htB);
    align_kernel  <<<dim3(NT / 32, NB),               256, 0, stream>>>(htB, mB, lens, out_align, Pb);
    ctx_kernel    <<<dim3(NT / 128, ND / 128, NB),    256, 0, stream>>>(Pb, mB, cB);
    outproj_kernel<<<dim3((NB * NT) / 128, ND / 128), 256, 0, stream>>>(cB, qB, wOutB, out_attn);
}
